// AuroraAttention_5935644803182
// MI455X (gfx1250) — matrix operand order repaired and checked by reading, not compiled, not run on hardware
//
#include <hip/hip_runtime.h>

// ---------------------------------------------------------------------------
// Types for CDNA5 WMMA (gfx1250, wave32)
// ---------------------------------------------------------------------------
typedef __bf16 bf16_t;
typedef __attribute__((ext_vector_type(16))) __bf16 v16bf;
typedef __attribute__((ext_vector_type(4)))  __bf16 v4bf;
typedef __attribute__((ext_vector_type(8)))  float  v8f;
typedef __attribute__((ext_vector_type(4)))  float  v4f;
typedef __attribute__((ext_vector_type(4)))  int    v4i;

union FragB {
    v16bf v;
    v4i   i[2];
};

__device__ __forceinline__ v8f wmma_bf16(v16bf a, v16bf b, v8f c) {
    // D = A(16x32) * B(32x16) + C, f32 accumulate
    return __builtin_amdgcn_wmma_f32_16x16x32_bf16(
        /*neg_a=*/false, a, /*neg_b=*/false, b,
        /*c_mod=*/(short)0, c, /*reuse_a=*/false, /*reuse_b=*/false);
}

// Low 32 bits of a generic pointer to __shared__ = LDS byte address
// (ISA: flat->LDS mapping uses addr[31:0] directly).
__device__ __forceinline__ unsigned int lds_addr_of(const void* p) {
    return (unsigned int)(unsigned long long)p;
}

// Async global -> LDS copy of 32 bytes (two b128 transfers), ASYNCcnt-tracked.
__device__ __forceinline__ void async_copy_32B(unsigned int lds_byte_addr,
                                               const bf16_t* gptr) {
    asm volatile("global_load_async_to_lds_b128 %0, %1, off"
                 :: "v"(lds_byte_addr), "v"(gptr) : "memory");
    asm volatile("global_load_async_to_lds_b128 %0, %1, off offset:16"
                 :: "v"(lds_byte_addr), "v"(gptr) : "memory");
}

__device__ __forceinline__ void wait_async0() {
    asm volatile("s_wait_asynccnt 0x0" ::: "memory");
}

// Problem constants
constexpr int Ltok = 4096;   // sequence length
constexpr int Dm   = 1024;   // embed dim
constexpr int H    = 16;     // heads
constexpr int HD   = 64;     // head dim
constexpr float QSCALE = 0.125f;  // 64^-0.5

// ---------------------------------------------------------------------------
// fp32 -> bf16 conversion (vectorized x4)
// ---------------------------------------------------------------------------
__global__ __launch_bounds__(256) void cvt_f32_bf16_kernel(
    const float* __restrict__ src, bf16_t* __restrict__ dst, int n4)
{
    int i = blockIdx.x * 256 + threadIdx.x;
    if (i < n4) {
        v4f f = *(const v4f*)(src + (size_t)i * 4);
        v4bf o;
        o[0] = (bf16_t)f[0]; o[1] = (bf16_t)f[1];
        o[2] = (bf16_t)f[2]; o[3] = (bf16_t)f[3];
        *(v4bf*)(dst + (size_t)i * 4) = o;
    }
}

// log(size) table for the attention bias
__global__ __launch_bounds__(256) void logsize_kernel(
    const float* __restrict__ size, float* __restrict__ out, int n)
{
    int i = blockIdx.x * 256 + threadIdx.x;
    if (i < n) out[i] = __logf(size[i]);
}

// metric[t,d] = mean over heads of K[t, h*64+d]
__global__ __launch_bounds__(256) void metric_kernel(
    const bf16_t* __restrict__ Kb, float* __restrict__ out)
{
    int i = blockIdx.x * 256 + threadIdx.x;      // 0 .. L*HD
    int t = i >> 6, d = i & 63;
    float s = 0.f;
    #pragma unroll
    for (int h = 0; h < H; ++h)
        s += (float)Kb[(size_t)t * Dm + h * HD + d];
    out[i] = s * (1.0f / 16.0f);
}

// ---------------------------------------------------------------------------
// GEMM: C = A(MxK) * W(NxK)^T + bias  with bf16 WMMA, f32 accumulate.
// Block tile 128x128, 8 wave32 waves (4 along M, 2 along N).
// K tiles staged into double-buffered LDS with GLOBAL_LOAD_ASYNC_TO_LDS_B128
// (ASYNCcnt), prefetching step k+32 while step k computes.
// mode 0: bf16 row-major out, scaled        (Q / K projections)
// mode 1: bf16 transposed out  out[n*M+m]   (V^T for attention)
// mode 2: f32 row-major out                 (final Wo projection)
// ---------------------------------------------------------------------------
#define G_LDSS 40   // LDS row stride (elems) = 80B = odd multiple of 16B

__global__ __launch_bounds__(256) void gemm_bf16_kernel(
    const bf16_t* __restrict__ A, const bf16_t* __restrict__ W,
    const float* __restrict__ bias, void* __restrict__ out,
    int M, int N, int K, int mode, float scale)
{
    __shared__ __align__(16) bf16_t lA[2][128 * G_LDSS];
    __shared__ __align__(16) bf16_t lB[2][128 * G_LDSS];

    const int tid  = threadIdx.x;
    const int lane = tid & 31;
    const int wid  = tid >> 5;
    const int wm   = wid & 3;     // wave M offset: 32 rows each
    const int wn   = wid >> 2;    // wave N offset: 64 cols each
    const int m0   = blockIdx.y * 128;
    const int n0   = blockIdx.x * 128;

    const int srow  = tid >> 1;   // staging: 128 rows x 2 halves of 16 elems
    const int shalf = tid & 1;

    const int l16    = lane & 15;
    const int halfHi = lane >> 4;  // 0/1

    // staging source/destination addresses
    const bf16_t* gAbase = A + (size_t)(m0 + srow) * K + shalf * 16;
    const bf16_t* gBbase = W + (size_t)(n0 + srow) * K + shalf * 16;
    unsigned int laA[2], laB[2];
    #pragma unroll
    for (int b = 0; b < 2; ++b) {
        laA[b] = lds_addr_of(&lA[b][srow * G_LDSS + shalf * 16]);
        laB[b] = lds_addr_of(&lB[b][srow * G_LDSS + shalf * 16]);
    }

    auto stage = [&](int buf, int k0) {
        async_copy_32B(laA[buf], gAbase + k0);
        async_copy_32B(laB[buf], gBbase + k0);
    };

    v8f acc[2][4];
    #pragma unroll
    for (int i = 0; i < 2; ++i)
        #pragma unroll
        for (int j = 0; j < 4; ++j)
            acc[i][j] = (v8f)0.0f;

    stage(0, 0);

    for (int k0 = 0; k0 < K; k0 += 32) {
        const int buf = (k0 >> 5) & 1;
        wait_async0();        // my async writes for this buffer are done
        __syncthreads();      // everyone's are done; prev-step reads retired
        if (k0 + 32 < K) stage(buf ^ 1, k0 + 32);   // prefetch next K tile

        // A fragments: lane holds row M=l16, K chunks {ab..ab+7, ab+16..ab+23}
        const int ab = halfHi ? 8 : 0;
        FragB af[2];
        #pragma unroll
        for (int i = 0; i < 2; ++i) {
            const bf16_t* p = &lA[buf][(wm * 32 + i * 16 + l16) * G_LDSS + ab];
            af[i].i[0] = *(const v4i*)(p);
            af[i].i[1] = *(const v4i*)(p + 16);
        }
        // B fragments: lane holds col N=l16, K chunks {bb..bb+7, bb+16..bb+23}
        const int bb = halfHi ? 8 : 0;
        FragB bfb[4];
        #pragma unroll
        for (int j = 0; j < 4; ++j) {
            const bf16_t* p = &lB[buf][(wn * 64 + j * 16 + l16) * G_LDSS + bb];
            bfb[j].i[0] = *(const v4i*)(p);
            bfb[j].i[1] = *(const v4i*)(p + 16);
        }
        #pragma unroll
        for (int i = 0; i < 2; ++i)
            #pragma unroll
            for (int j = 0; j < 4; ++j)
                acc[i][j] = wmma_bf16(af[i].v, bfb[j].v, acc[i][j]);
    }

    // epilogue: C layout lane holds rows r + 8*halfHi, col = l16
    #pragma unroll
    for (int j = 0; j < 4; ++j) {
        const int gn = n0 + wn * 64 + j * 16 + l16;
        const float bv = bias[gn];
        #pragma unroll
        for (int i = 0; i < 2; ++i) {
            #pragma unroll
            for (int r = 0; r < 8; ++r) {
                const int gm = m0 + wm * 32 + i * 16 + halfHi * 8 + r;
                const float c = (acc[i][j][r] + bv) * scale;
                if (mode == 0)
                    ((bf16_t*)out)[(size_t)gm * N + gn] = (bf16_t)c;
                else if (mode == 1)
                    ((bf16_t*)out)[(size_t)gn * M + gm] = (bf16_t)c;
                else
                    ((float*)out)[(size_t)gm * N + gn] = c;
            }
        }
    }
}

// ---------------------------------------------------------------------------
// Flash attention: block = (64 queries, 1 head), 4 waves x 16 queries.
// Per 64-key block: S = Q K^T (8 wmma) + log(size) bias, online softmax
// with half-wave shuffle reductions, P routed via padded LDS to A-layout,
// O += P V (8 wmma).  Q is pre-scaled; V stored transposed (h, hd, t).
// ---------------------------------------------------------------------------
#define P_LDSS 88   // P LDS row stride (elems) = 176B = odd multiple of 16B

__global__ __launch_bounds__(128) void attn_kernel(
    const bf16_t* __restrict__ Qb,   // L x D
    const bf16_t* __restrict__ Kb,   // L x D
    const bf16_t* __restrict__ Vt,   // (h*64+d) x L
    const float*  __restrict__ lsz,  // L (log sizes)
    bf16_t* __restrict__ O)          // L x D
{
    __shared__ __align__(16) bf16_t pLds[4][16 * P_LDSS];

    const int tid    = threadIdx.x;
    const int lane   = tid & 31;
    const int w      = tid >> 5;
    const int l16    = lane & 15;
    const int halfHi = lane >> 4;
    const int h      = blockIdx.y;
    const int q0     = blockIdx.x * 64 + w * 16;

    // Q fragments (A operand, M=16 queries, K=hd, 2 k-steps), kept in regs
    FragB qf[2];
    {
        const int ab = halfHi ? 8 : 0;
        #pragma unroll
        for (int ks = 0; ks < 2; ++ks) {
            const bf16_t* p = Qb + (size_t)(q0 + l16) * Dm + h * HD + ks * 32 + ab;
            qf[ks].i[0] = *(const v4i*)(p);
            qf[ks].i[1] = *(const v4i*)(p + 16);
        }
    }

    v8f o_acc[4];
    float m_r[8], l_r[8];
    #pragma unroll
    for (int j = 0; j < 4; ++j) o_acc[j] = (v8f)0.0f;
    #pragma unroll
    for (int r = 0; r < 8; ++r) { m_r[r] = -3.0e38f; l_r[r] = 0.0f; }

    for (int kb = 0; kb < Ltok; kb += 64) {
        // ---- S = Q K^T over 4 key tiles of 16 ----
        v8f s[4];
        #pragma unroll
        for (int j = 0; j < 4; ++j) s[j] = (v8f)0.0f;
        const int bb = halfHi ? 8 : 0;
        #pragma unroll
        for (int j = 0; j < 4; ++j) {
            #pragma unroll
            for (int ks = 0; ks < 2; ++ks) {
                const bf16_t* p =
                    Kb + (size_t)(kb + j * 16 + l16) * Dm + h * HD + ks * 32 + bb;
                FragB kf;
                kf.i[0] = *(const v4i*)(p);
                kf.i[1] = *(const v4i*)(p + 16);
                s[j] = wmma_bf16(qf[ks].v, kf.v, s[j]);
            }
        }

        // ---- bias + row max ----
        float rowmax[8];
        #pragma unroll
        for (int r = 0; r < 8; ++r) rowmax[r] = -3.0e38f;
        #pragma unroll
        for (int j = 0; j < 4; ++j) {
            const float bias = lsz[kb + j * 16 + l16];
            #pragma unroll
            for (int r = 0; r < 8; ++r) {
                const float x = s[j][r] + bias;
                s[j][r] = x;
                rowmax[r] = fmaxf(rowmax[r], x);
            }
        }
        #pragma unroll
        for (int r = 0; r < 8; ++r)
            #pragma unroll
            for (int d = 1; d < 16; d <<= 1)
                rowmax[r] = fmaxf(rowmax[r], __shfl_xor(rowmax[r], d, 32));

        float corr[8], rowsum[8];
        #pragma unroll
        for (int r = 0; r < 8; ++r) {
            const float mnew = fmaxf(m_r[r], rowmax[r]);
            corr[r] = __expf(m_r[r] - mnew);
            m_r[r] = mnew;
            rowsum[r] = 0.0f;
        }

        // ---- P = exp(S - m), write to LDS in (row, key) layout ----
        #pragma unroll
        for (int j = 0; j < 4; ++j) {
            #pragma unroll
            for (int r = 0; r < 8; ++r) {
                const float p = __expf(s[j][r] - m_r[r]);
                rowsum[r] += p;
                pLds[w][(halfHi * 8 + r) * P_LDSS + j * 16 + l16] = (bf16_t)p;
            }
        }
        #pragma unroll
        for (int r = 0; r < 8; ++r) {
            #pragma unroll
            for (int d = 1; d < 16; d <<= 1)
                rowsum[r] += __shfl_xor(rowsum[r], d, 32);
            l_r[r] = l_r[r] * corr[r] + rowsum[r];
        }
        #pragma unroll
        for (int j = 0; j < 4; ++j)
            #pragma unroll
            for (int r = 0; r < 8; ++r)
                o_acc[j][r] *= corr[r];

        // ---- O += P V  (A = P from LDS, B = V^T rows, K-contiguous) ----
        #pragma unroll
        for (int ks = 0; ks < 2; ++ks) {
            FragB pf;
            const bf16_t* pp = &pLds[w][l16 * P_LDSS + ks * 32 + halfHi * 8];
            pf.i[0] = *(const v4i*)(pp);
            pf.i[1] = *(const v4i*)(pp + 16);
            #pragma unroll
            for (int j = 0; j < 4; ++j) {
                const bf16_t* vp =
                    Vt + (size_t)(h * HD + j * 16 + l16) * Ltok + kb + ks * 32 + bb;
                FragB vf;
                vf.i[0] = *(const v4i*)(vp);
                vf.i[1] = *(const v4i*)(vp + 16);
                o_acc[j] = wmma_bf16(pf.v, vf.v, o_acc[j]);
            }
        }
    }

    // normalize and store to attn buffer (t, h*64+d) as bf16
    #pragma unroll
    for (int j = 0; j < 4; ++j) {
        const int gn = h * HD + j * 16 + l16;
        #pragma unroll
        for (int r = 0; r < 8; ++r) {
            const int gm = q0 + halfHi * 8 + r;
            O[(size_t)gm * Dm + gn] = (bf16_t)(o_acc[j][r] / l_r[r]);
        }
    }
}

// ---------------------------------------------------------------------------
// Host-side launch
// ---------------------------------------------------------------------------
extern "C" void kernel_launch(void* const* d_in, const int* in_sizes, int n_in,
                              void* d_out, int out_size, void* d_ws, size_t ws_size,
                              hipStream_t stream) {
    const float* X   = (const float*)d_in[0];
    const float* sz  = (const float*)d_in[1];
    const float* wq  = (const float*)d_in[2];
    const float* bq  = (const float*)d_in[3];
    const float* wk  = (const float*)d_in[4];
    const float* bk  = (const float*)d_in[5];
    const float* wv  = (const float*)d_in[6];
    const float* bv  = (const float*)d_in[7];
    const float* wo  = (const float*)d_in[8];
    const float* bo  = (const float*)d_in[9];

    const size_t LD = (size_t)Ltok * Dm;   // 4M elems
    const size_t DD = (size_t)Dm * Dm;     // 1M elems

    bf16_t* ws   = (bf16_t*)d_ws;
    bf16_t* Xb   = ws;
    bf16_t* Qb   = ws + 1 * LD;
    bf16_t* Kb   = ws + 2 * LD;
    bf16_t* Vtb  = ws + 3 * LD;
    bf16_t* Ab   = ws + 4 * LD;           // attention output bf16
    bf16_t* Wqb  = ws + 5 * LD;
    bf16_t* Wkb  = Wqb + DD;
    bf16_t* Wvb  = Wkb + DD;
    bf16_t* Wob  = Wvb + DD;
    float*  lsz  = (float*)(Wob + DD);

    float* out    = (float*)d_out;          // (L, D)
    float* metric = out + LD;               // (L, HD)

    // fp32 -> bf16 conversions + log(size)
    cvt_f32_bf16_kernel<<<(int)(LD / 1024), 256, 0, stream>>>(X,  Xb,  (int)(LD / 4));
    cvt_f32_bf16_kernel<<<(int)(DD / 1024), 256, 0, stream>>>(wq, Wqb, (int)(DD / 4));
    cvt_f32_bf16_kernel<<<(int)(DD / 1024), 256, 0, stream>>>(wk, Wkb, (int)(DD / 4));
    cvt_f32_bf16_kernel<<<(int)(DD / 1024), 256, 0, stream>>>(wv, Wvb, (int)(DD / 4));
    cvt_f32_bf16_kernel<<<(int)(DD / 1024), 256, 0, stream>>>(wo, Wob, (int)(DD / 4));
    logsize_kernel<<<Ltok / 256, 256, 0, stream>>>(sz, lsz, Ltok);

    // QKV projections (Q pre-scaled; V written transposed per head-dim)
    dim3 ggrid(Dm / 128, Ltok / 128);
    gemm_bf16_kernel<<<ggrid, 256, 0, stream>>>(Xb, Wqb, bq, Qb,  Ltok, Dm, Dm, 0, QSCALE);
    gemm_bf16_kernel<<<ggrid, 256, 0, stream>>>(Xb, Wkb, bk, Kb,  Ltok, Dm, Dm, 0, 1.0f);
    gemm_bf16_kernel<<<ggrid, 256, 0, stream>>>(Xb, Wvb, bv, Vtb, Ltok, Dm, Dm, 1, 1.0f);

    // metric = head-mean of K
    metric_kernel<<<(Ltok * HD) / 256, 256, 0, stream>>>(Kb, metric);

    // flash attention
    attn_kernel<<<dim3(Ltok / 64, H), 128, 0, stream>>>(Qb, Kb, Vtb, lsz, Ab);

    // output projection (fp32 out)
    gemm_bf16_kernel<<<ggrid, 256, 0, stream>>>(Ab, Wob, bo, out, Ltok, Dm, Dm, 2, 1.0f);
}
